// MHSAModule_17849884082614
// MI455X (gfx1250) — matrix operand order repaired and checked by reading, not compiled, not run on hardware
//
#include <hip/hip_runtime.h>

namespace {

constexpr int NB = 8, NT = 1024, ND = 512, NH = 8, NHS = 64;
constexpr int NTOK = NB * NT;

typedef __attribute__((ext_vector_type(16))) __bf16 v16bf;
typedef __attribute__((ext_vector_type(8)))  float  v8f;

__device__ __forceinline__ v8f wmma_bf(v16bf a, v16bf b, v8f c) {
  // D = A(16x32 bf16) * B(32x16 bf16) + C(16x16 f32)
  return __builtin_amdgcn_wmma_f32_16x16x32_bf16(false, a, false, b, (short)0, c,
                                                 false, false);
}

// ---- WMMA fragment layouts (CDNA5 ISA 7.12.2) -------------------------------
// A 16x32 bf16: lane L holds row M=L&15; half e -> K = e + (e>=8?8:0) + (L>=16?8:0)
// B 32x16 bf16: lane L holds col N=L&15; half e -> K = e + (e>=8?8:0) + (L>=16?8:0)
// C  f32 16x16: lane L, reg r -> (M = r + 8*(L>=16), N = L&15)

// A fragment from row-major [*, lda] (two contiguous 16B runs per lane).
__device__ __forceinline__ v16bf load_a(const __bf16* tile, int lda, int lane) {
  v16bf f;
  int m = lane & 15;
  int kh = (lane >= 16) ? 8 : 0;
#pragma unroll
  for (int e = 0; e < 16; ++e) {
    int k = e + ((e >= 8) ? 8 : 0) + kh;
    f[e] = tile[(size_t)m * lda + k];
  }
  return f;
}

// A fragment from [T][NHS] at absolute row r0+m, clamped to T-1.
__device__ __forceinline__ v16bf load_a_clamp(const __bf16* arr, int r0, int k0,
                                              int lane) {
  v16bf f;
  int m = lane & 15;
  int kh = (lane >= 16) ? 16 : 0;
  int r = r0 + m;
  if (r > NT - 1) r = NT - 1;
#pragma unroll
  for (int e = 0; e < 16; ++e) {
    int k = e + kh;
    f[e] = arr[(size_t)r * NHS + k0 + k];
  }
  return f;
}

// B fragment from pre-swizzled tile store: one aligned 32B vector load per lane.
// Swizzled layout: ((kt*ntiles_n + ct)*32 + lane)*16 + e
__device__ __forceinline__ v16bf load_b_swz(const __bf16* base, int kt, int ct,
                                            int ntiles_n, int lane) {
  return *(const v16bf*)(base + ((((size_t)kt * ntiles_n + ct) * 32 + lane) << 4));
}

// B = rows^T for q.k^T style: B[k][n] = rows[n][ks+k] (contiguous 32B per lane).
__device__ __forceinline__ v16bf load_bT(const __bf16* rows, int ks, int lane) {
  v16bf f;
  int n = lane & 15;
  int kh = (lane >= 16) ? 16 : 0;
#pragma unroll
  for (int e = 0; e < 16; ++e) f[e] = rows[(size_t)n * NHS + ks + kh + e];
  return f;
}

// Windowed p^T fragment (rel-shift window): B[k][n] = p[r0+n][ks+k], zero if OOB.
__device__ __forceinline__ v16bf load_pT(const __bf16* p, int r0, int ks, int lane) {
  v16bf f;
  int n = lane & 15;
  int kh = (lane >= 16) ? 16 : 0;
  int r = r0 + n;
  if (r >= 0 && r < NT) {
#pragma unroll
    for (int e = 0; e < 16; ++e) f[e] = p[(size_t)r * NHS + ks + kh + e];
  } else {
#pragma unroll
    for (int e = 0; e < 16; ++e) f[e] = (__bf16)0.0f;
  }
  return f;
}

// B for probs*V from 16x16-tile-swizzled V; only K=0..15 valid (upper half zero
// to match the zero-padded probs A fragment).
__device__ __forceinline__ v16bf load_vB_swz(const __bf16* vswz, int rt, int f,
                                             int lane) {
  if (lane < 16)
    return *(const v16bf*)(vswz + ((((size_t)rt * 4 + f) * 16 + lane) << 4));
  v16bf z;
#pragma unroll
  for (int e = 0; e < 16; ++e) z[e] = (__bf16)0.0f;
  return z;
}

} // namespace

// ---- LayerNorm (pre-norm) -> bf16 ------------------------------------------
__global__ void ln_kernel(const float* __restrict__ x, const float* __restrict__ gamma,
                          const float* __restrict__ beta, __bf16* __restrict__ y) {
  __shared__ float red[256];
  int row = blockIdx.x;
  int tid = threadIdx.x;
  const float* xr = x + (size_t)row * ND;
  float v0 = xr[tid], v1 = xr[tid + 256];
  red[tid] = v0 + v1;
  __syncthreads();
  for (int o = 128; o > 0; o >>= 1) {
    if (tid < o) red[tid] += red[tid + o];
    __syncthreads();
  }
  float mu = red[0] * (1.0f / ND);
  __syncthreads();
  float d0 = v0 - mu, d1 = v1 - mu;
  red[tid] = d0 * d0 + d1 * d1;
  __syncthreads();
  for (int o = 128; o > 0; o >>= 1) {
    if (tid < o) red[tid] += red[tid + o];
    __syncthreads();
  }
  float rs = rsqrtf(red[0] * (1.0f / ND) + 1e-3f);
  __bf16* yr = y + (size_t)row * ND;
  yr[tid] = (__bf16)(d0 * rs * gamma[tid] + beta[tid]);
  yr[tid + 256] = (__bf16)(d1 * rs * gamma[tid + 256] + beta[tid + 256]);
}

// ---- fp32 -> bf16 plain convert (pos) ---------------------------------------
__global__ void cvt_kernel(const float* __restrict__ in, __bf16* __restrict__ out,
                           int n) {
  int i = blockIdx.x * blockDim.x + threadIdx.x;
  if (i < n) out[i] = (__bf16)in[i];
}

// ---- fp32 [K][N] -> bf16 B-fragment-swizzled tiles --------------------------
// out[mi][((kt*(N/16)+ct)*32+l)*16+e] = in[mi][(kt*32 + (e>=8?16:0) + (l>=16?8:0) + (e&7))*N + ct*16 + (l&15)]
__global__ void cvtw_kernel(const float* __restrict__ in, __bf16* __restrict__ out,
                            int K, int N, int nmat) {
  int i = blockIdx.x * blockDim.x + threadIdx.x;
  int per = K * N;
  if (i >= per * nmat) return;
  int mi = i / per;
  int o = i - mi * per;
  int e = o & 15;
  int l = (o >> 4) & 31;
  int tile = o >> 9;
  int nt = N >> 4;
  int ct = tile % nt;
  int kt = tile / nt;
  int k = kt * 32 + ((e >= 8) ? 16 : 0) + ((l >= 16) ? 8 : 0) + (e & 7);
  int n = ct * 16 + (l & 15);
  out[(size_t)mi * per + o] = (__bf16)in[(size_t)mi * per + (size_t)k * N + n];
}

// ---- Per-head projections: q(+u,+v), k, v, p  (wave w = projection type) ----
__global__ void proj_kernel(const __bf16* __restrict__ yb, const __bf16* __restrict__ posb,
                            const __bf16* __restrict__ wqb, const __bf16* __restrict__ wkb,
                            const __bf16* __restrict__ wvb, const __bf16* __restrict__ wpb,
                            const float* __restrict__ bu, const float* __restrict__ bv,
                            __bf16* __restrict__ qu, __bf16* __restrict__ qv,
                            __bf16* __restrict__ kk, __bf16* __restrict__ vv,
                            __bf16* __restrict__ pp) {
  int h = blockIdx.y;
  int row0 = blockIdx.x * 16;
  int wave = threadIdx.x >> 5;
  int lane = threadIdx.x & 31;
  const __bf16* A = (wave == 3) ? posb : yb;
  const __bf16* W = (wave == 0) ? wqb : (wave == 1) ? wkb : (wave == 2) ? wvb : wpb;
  const __bf16* Bm = W + (size_t)h * ND * NHS;  // swizzled per-head block

  v8f acc[4] = {};
  for (int k0 = 0; k0 < ND; k0 += 32) {
    v16bf af = load_a(A + (size_t)row0 * ND + k0, ND, lane);
#pragma unroll
    for (int f = 0; f < 4; ++f)
      acc[f] = wmma_bf(af, load_b_swz(Bm, k0 >> 5, f, 4, lane), acc[f]);
  }

  int Mh = (lane >> 4) * 8, N0 = lane & 15;
#pragma unroll
  for (int f = 0; f < 4; ++f) {
#pragma unroll
    for (int r = 0; r < 8; ++r) {
      int row = row0 + Mh + r;
      int b = row >> 10, t = row & (NT - 1);
      int n = f * 16 + N0;
      size_t bh = (((size_t)b * NH + h) * NT) * NHS;
      float val = acc[f][r];
      if (wave == 0) {
        size_t o = bh + (size_t)t * NHS + n;
        qu[o] = (__bf16)(val + bu[h * NHS + n]);
        qv[o] = (__bf16)(val + bv[h * NHS + n]);
      } else if (wave == 1) {
        kk[bh + (size_t)t * NHS + n] = (__bf16)val;
      } else if (wave == 2) {
        // V in 16x16-tile swizzled layout for attn's probs*V B fragments
        vv[bh + ((((size_t)(t >> 4) * 4 + f) * 16 + N0) << 4) + (t & 15)] = (__bf16)val;
      } else {
        pp[bh + (size_t)t * NHS + n] = (__bf16)val;
      }
    }
  }
}

// ---- Fused flash attention with exact Transformer-XL rel-shift --------------
// bd[t,m] = s[t, m-t+T-1]  if m <= t;  0 if m == t+1;  s[t+1, m-t-2] if m > t+1
// where s[t,j] = (q[t]+v_bias).p[j].  Per 16x16 tile the shifted entries lie on
// diagonals of two 16x32 windowed WMMA products W1 (rows t) / W2 (rows t+1).
__global__ void attn_kernel(const __bf16* __restrict__ qu, const __bf16* __restrict__ qv,
                            const __bf16* __restrict__ kk, const __bf16* __restrict__ vv,
                            const __bf16* __restrict__ pp, __bf16* __restrict__ ctxswz) {
  __shared__ float ldsW1[4][16][32];
  __shared__ float ldsW2[4][16][32];
  __shared__ float ldsP[4][16][16];
  int wave = threadIdx.x >> 5, lane = threadIdx.x & 31;
  int b = blockIdx.z, h = blockIdx.y;
  int t0 = blockIdx.x * 64 + wave * 16;
  float (*W1)[32] = ldsW1[wave];
  float (*W2)[32] = ldsW2[wave];
  float (*PB)[16] = ldsP[wave];

  size_t base = (((size_t)b * NH + h) * NT) * NHS;
  const __bf16* QU = qu + base;
  const __bf16* QV = qv + base;
  const __bf16* K = kk + base;
  const __bf16* V = vv + base;  // swizzled
  const __bf16* P = pp + base;

  v16bf quA[2], qvA[2], qsA[2];
#pragma unroll
  for (int ks = 0; ks < 2; ++ks) {
    quA[ks] = load_a_clamp(QU, t0, ks * 32, lane);
    qvA[ks] = load_a_clamp(QV, t0, ks * 32, lane);
    qsA[ks] = load_a_clamp(QV, t0 + 1, ks * 32, lane);  // rows shifted by +1
  }

  int Mh = (lane >> 4) * 8, N0 = lane & 15;
  float mrow[8], lrow[8];
  v8f ctx[4] = {};
#pragma unroll
  for (int r = 0; r < 8; ++r) {
    mrow[r] = -1e30f;
    lrow[r] = 0.f;
  }

  for (int m0 = 0; m0 < NT; m0 += 16) {
    if (m0 + 16 < NT) {  // prefetch next K/V tiles into cache (global_prefetch_b8)
      __builtin_prefetch(K + (size_t)(m0 + 16) * NHS, 0, 1);
      __builtin_prefetch(V + ((size_t)((m0 >> 4) + 1) * 4) * 256, 0, 1);
    }

    // content scores ac = (q+u) . k^T  (K = 64 -> two WMMA k-steps)
    v8f ac = {};
#pragma unroll
    for (int ks = 0; ks < 2; ++ks)
      ac = wmma_bf(quA[ks], load_bT(K + (size_t)m0 * NHS, ks * 32, lane), ac);

    // windowed position scores: col c -> p-row base+c
    int pb1 = m0 - t0 + NT - 17;  // W1[M][16 + (N-M)] = s[t, m-t+T-1]
    int pb2 = m0 - t0 - 18;       // W2[M][16 + (N-M)] = s[t+1, m-t-2]
    v8f w1[2] = {}, w2[2] = {};
#pragma unroll
    for (int nt = 0; nt < 2; ++nt) {
#pragma unroll
      for (int ks = 0; ks < 2; ++ks) {
        w1[nt] = wmma_bf(qvA[ks], load_pT(P, pb1 + nt * 16, ks * 32, lane), w1[nt]);
        w2[nt] = wmma_bf(qsA[ks], load_pT(P, pb2 + nt * 16, ks * 32, lane), w2[nt]);
      }
    }
    // spill windows to this wave's LDS for diagonal gather (same-wave DS in-order)
#pragma unroll
    for (int nt = 0; nt < 2; ++nt) {
#pragma unroll
      for (int r = 0; r < 8; ++r) {
        W1[Mh + r][nt * 16 + N0] = w1[nt][r];
        W2[Mh + r][nt * 16 + N0] = w2[nt][r];
      }
    }

    // logits + online softmax (row data lives in a 16-lane half)
    float pv[8];
#pragma unroll
    for (int r = 0; r < 8; ++r) {
      int M = Mh + r;
      int t = t0 + M, m = m0 + N0;
      int c = 16 + N0 - M;  // in [1,31]
      float bd = (m <= t) ? W1[M][c] : ((m == t + 1) ? 0.f : W2[M][c]);
      float lg = (ac[r] + bd) * 0.125f;  // / sqrt(HS)
      float mx = lg;
#pragma unroll
      for (int o = 1; o < 16; o <<= 1) mx = fmaxf(mx, __shfl_xor(mx, o, 32));
      float nm = fmaxf(mrow[r], mx);
      float sc = __expf(mrow[r] - nm);
      float pe = __expf(lg - nm);
      float sm = pe;
#pragma unroll
      for (int o = 1; o < 16; o <<= 1) sm += __shfl_xor(sm, o, 32);
      lrow[r] = lrow[r] * sc + sm;
      mrow[r] = nm;
      pv[r] = pe;
#pragma unroll
      for (int f = 0; f < 4; ++f) ctx[f][r] *= sc;
    }

    // probs tile -> LDS -> A fragment (K padded to 32 with zeros)
#pragma unroll
    for (int r = 0; r < 8; ++r) PB[Mh + r][N0] = pv[r];
    v16bf pA;
    {
      int m = lane & 15;
      int kh = (lane >= 16) ? 16 : 0;
#pragma unroll
      for (int e = 0; e < 16; ++e) {
        int k = e + kh;
        pA[e] = (k < 16) ? (__bf16)PB[m][k] : (__bf16)0.0f;
      }
    }
#pragma unroll
    for (int f = 0; f < 4; ++f)
      ctx[f] = wmma_bf(pA, load_vB_swz(V, m0 >> 4, f, lane), ctx[f]);
  }

  // normalize and store ctx directly in bf16 A-fragment-swizzled layout
  // (rows = b*T+t, K-dim = h*64+hs) so outproj's A loads are aligned vectors.
  int rt = (b * NT + t0) >> 4;
#pragma unroll
  for (int f = 0; f < 4; ++f) {
#pragma unroll
    for (int r = 0; r < 8; ++r) {
      int M = Mh + r;
      int col = h * NHS + f * 16 + N0;
      int kt = col >> 5, kr = col & 31;
      int hi = (kr >> 3) & 1;                       // which lane-half owns this K
      int lt = M + (hi ? 16 : 0);                   // A-frag lane
      int e = (kr & 7) + ((kr >= 16) ? 8 : 0);      // A-frag half index
      ctxswz[((((size_t)rt * 16 + kt) * 32 + lt) << 4) + e] =
          (__bf16)(ctx[f][r] / lrow[r]);
    }
  }
}

// ---- Output projection + bias + residual ------------------------------------
__global__ void outproj_kernel(const __bf16* __restrict__ ctxswz,
                               const __bf16* __restrict__ wob,
                               const float* __restrict__ bo, const float* __restrict__ x,
                               float* __restrict__ out) {
  int wave = threadIdx.x >> 5, lane = threadIdx.x & 31;
  int row0 = blockIdx.x * 64 + wave * 16;
  int col0 = blockIdx.y * 64;
  int rt = row0 >> 4;
  v8f acc[4] = {};
  for (int k0 = 0; k0 < ND; k0 += 32) {
    int kt = k0 >> 5;
    v16bf af = *(const v16bf*)(ctxswz + ((((size_t)rt * 16 + kt) * 32 + lane) << 4));
#pragma unroll
    for (int f = 0; f < 4; ++f)
      acc[f] = wmma_bf(af, load_b_swz(wob, kt, (col0 >> 4) + f, 32, lane), acc[f]);
  }
  int Mh = (lane >> 4) * 8, N0 = lane & 15;
#pragma unroll
  for (int f = 0; f < 4; ++f) {
#pragma unroll
    for (int r = 0; r < 8; ++r) {
      int row = row0 + Mh + r, col = col0 + f * 16 + N0;
      size_t i = (size_t)row * ND + col;
      out[i] = acc[f][r] + bo[col] + x[i];
    }
  }
}

extern "C" void kernel_launch(void* const* d_in, const int* in_sizes, int n_in,
                              void* d_out, int out_size, void* d_ws, size_t ws_size,
                              hipStream_t stream) {
  (void)in_sizes; (void)n_in; (void)out_size; (void)ws_size;
  const float* x = (const float*)d_in[0];
  const float* pos = (const float*)d_in[1];
  const float* gamma = (const float*)d_in[2];
  const float* beta = (const float*)d_in[3];
  const float* wq = (const float*)d_in[4];
  const float* wk = (const float*)d_in[5];
  const float* wv = (const float*)d_in[6];
  const float* wpos = (const float*)d_in[7];
  const float* bu = (const float*)d_in[8];
  const float* bv = (const float*)d_in[9];
  const float* wo = (const float*)d_in[10];
  const float* bo = (const float*)d_in[11];
  float* out = (float*)d_out;

  // workspace carve-up (~67 MB total)
  char* p = (char*)d_ws;
  auto take = [&](size_t bytes) {
    char* q = p;
    p += (bytes + 255) & ~(size_t)255;
    return q;
  };
  __bf16* yb = (__bf16*)take((size_t)NTOK * ND * 2);
  __bf16* posb = (__bf16*)take((size_t)NTOK * ND * 2);
  __bf16* wqb = (__bf16*)take((size_t)NH * ND * NHS * 2);
  __bf16* wkb = (__bf16*)take((size_t)NH * ND * NHS * 2);
  __bf16* wvb = (__bf16*)take((size_t)NH * ND * NHS * 2);
  __bf16* wpb = (__bf16*)take((size_t)NH * ND * NHS * 2);
  __bf16* wob = (__bf16*)take((size_t)ND * ND * 2);
  size_t nqe = (size_t)NB * NH * NT * NHS;
  __bf16* quB = (__bf16*)take(nqe * 2);
  __bf16* qvB = (__bf16*)take(nqe * 2);
  __bf16* kkB = (__bf16*)take(nqe * 2);
  __bf16* vvB = (__bf16*)take(nqe * 2);
  __bf16* ppB = (__bf16*)take(nqe * 2);
  __bf16* ctxswz = (__bf16*)take((size_t)NTOK * ND * 2);

  ln_kernel<<<NTOK, 256, 0, stream>>>(x, gamma, beta, yb);

  int npos = NTOK * ND;
  cvt_kernel<<<(npos + 255) / 256, 256, 0, stream>>>(pos, posb, npos);

  // per-head projection weights -> B-fragment swizzled bf16
  int nw = NH * ND * NHS;
  cvtw_kernel<<<(nw + 255) / 256, 256, 0, stream>>>(wq, wqb, ND, NHS, NH);
  cvtw_kernel<<<(nw + 255) / 256, 256, 0, stream>>>(wk, wkb, ND, NHS, NH);
  cvtw_kernel<<<(nw + 255) / 256, 256, 0, stream>>>(wv, wvb, ND, NHS, NH);
  cvtw_kernel<<<(nw + 255) / 256, 256, 0, stream>>>(wpos, wpb, ND, NHS, NH);
  int nwo = ND * ND;
  cvtw_kernel<<<(nwo + 255) / 256, 256, 0, stream>>>(wo, wob, ND, ND, 1);

  proj_kernel<<<dim3(NTOK / 16, NH), 128, 0, stream>>>(yb, posb, wqb, wkb, wvb, wpb,
                                                       bu, bv, quB, qvB, kkB, vvB, ppB);

  attn_kernel<<<dim3(NT / 64, NH, NB), 128, 0, stream>>>(quB, qvB, kkB, vvB, ppB,
                                                         ctxswz);

  outproj_kernel<<<dim3(NTOK / 64, ND / 64), 128, 0, stream>>>(ctxswz, wob, bo, x, out);
}
